// DiffusionModel_23665269801113
// MI455X (gfx1250) — compile-verified, matrix operand order checked by reading, not run on hardware
//
#include <hip/hip_runtime.h>
#include <stdint.h>

// ---------------------------------------------------------------------------
// Diffusion scan, MI455X (gfx1250).
//   x_{t+1} = x_t + BETA*( [x_t, t] @ W + b ) + sqrt(2 BETA)*noise
// [x,t]@W = x@W_top + t*colsum(W_bot); batch rows are independent, so one
// persistent kernel keeps each 64-row slab in WMMA C-fragments (VGPRs) for
// all 100 steps. GEMM = v_wmma_f32_16x16x32_f16; A staged through LDS in
// fragment order (base+immediate ds_load_b128); B streamed from L2 each step
// (asm-opaqued offset defeats LICM so nothing is hoisted/spilled to scratch).
// Noise = counter-based CLT gaussian (popcount) to keep VALU-per-step well
// under the WMMA latency budget so the matrix pipe stays the limiter.
// Roofline: 1.72 TFLOP total, ~130 MB HBM traffic -> matrix-engine bound.
// ---------------------------------------------------------------------------

#define BATCH      32768
#define DIM        512
#define KDIM       512
#define NUM_STEPS  100
#define BETA       0.01f
#define SIGMA      0.14142135623730951f  // sqrt(2*BETA)

#define M_TILE     64        // rows per workgroup
#define THREADS    512       // 16 waves: 4 row-strips x 4 col-quarters

typedef __attribute__((ext_vector_type(16))) _Float16 v16h;
typedef __attribute__((ext_vector_type(8)))  float    v8f;

union Frag { v16h v; uint4 q[2]; };

__device__ __forceinline__ uint32_t hashmix(uint32_t x) {
  x ^= x >> 16; x *= 0x7feb352du;
  x ^= x >> 15; x *= 0x846ca68bu;
  x ^= x >> 16;
  return x;
}

// Deterministic counter-based ~N(0,1): two 32-bit Binomial popcounts plus a
// continuous uniform dither. var = 8 + 8 + 1/12; cheap (no transcendentals),
// graph-replay safe.
__device__ __forceinline__ float gaussian(uint32_t seed) {
  uint32_t a = hashmix(seed);
  uint32_t c = hashmix(seed ^ 0x9e3779b9u);
  float d = (float)(__popc(a) + __popc(c) - 32);
  float u = (float)(a >> 9) * (1.0f / 8388608.0f) - 0.5f;
  return (d + u) * 0.24935132f;   // 1/sqrt(16 + 1/12)
}

// ---------------------------------------------------------------------------
// Prep 1: pack f16(W_top) in WMMA-B fragment order: fragment (ntile,ktile) is
// 1 KB, lane-major 32 B (two b128 loads per lane).
//   lanes 0-15 : n=lane,    halves 0..7 = K 0..7,  8..15 = K 16..23
//   lanes16-31 : n=lane-16, halves 0..7 = K 8..15, 8..15 = K 24..31
// ---------------------------------------------------------------------------
__global__ void pack_w_kernel(const float* __restrict__ W,
                              _Float16* __restrict__ Wpk) {
  int i = blockIdx.x * blockDim.x + threadIdx.x;   // 0 .. 512*512-1
  int k = i >> 9;
  int n = i & (DIM - 1);
  int kt = k >> 5, kl = k & 31;
  int ntg = n >> 4, nl = n & 15;
  int lane, h;
  if      (kl <  8) { lane = nl;      h = kl;      }
  else if (kl < 16) { lane = 16 + nl; h = kl - 8;  }
  else if (kl < 24) { lane = nl;      h = kl - 8;  }
  else              { lane = 16 + nl; h = kl - 16; }
  int idx = (((((ntg << 4) | kt) << 5) | lane) << 4) | h;
  Wpk[idx] = (_Float16)W[k * DIM + n];
}

// Prep 2: csum[n] = sum_k W[DIM + k][n]
__global__ void csum_kernel(const float* __restrict__ W,
                            float* __restrict__ csum) {
  int n = blockIdx.x * blockDim.x + threadIdx.x;
  if (n < DIM) {
    float s = 0.0f;
    for (int k = 0; k < KDIM; ++k) s += W[(KDIM + k) * DIM + n];
    csum[n] = s;
  }
}

// ---------------------------------------------------------------------------
// Main persistent kernel. Wave (s,q): s = 16-row strip, q = 128-col quarter.
// LDS holds f16(BETA*x_t) already in A-fragment order:
//   half-index = s*8192 + kt*512 + plane*256 + lane*8 + h      (64 KB total)
// ---------------------------------------------------------------------------
__global__ __launch_bounds__(THREADS)
void diffusion_kernel(const float* __restrict__ x0,
                      const _Float16* __restrict__ Wpk,
                      const float* __restrict__ b,
                      const float* __restrict__ csum,
                      float* __restrict__ out) {
  __shared__ _Float16 alds[4 * 16 * 512];    // 64 KB

  const int lane    = threadIdx.x & 31;
  const int wave    = threadIdx.x >> 5;
  const int s       = wave >> 2;
  const int q       = wave & 3;
  const int rowBase = blockIdx.x * M_TILE;
  const int mHi     = (lane >> 4) * 8;       // C layout: lanes 16-31 -> M+8
  const int nLane   = lane & 15;

  // Persistent x in C-fragments: xc[nt] = 16x16 tile at cols q*128 + nt*16
  v8f   xc[8];
  float bcol[8], ccol[8];
  #pragma unroll
  for (int nt = 0; nt < 8; ++nt) {
    int col  = q * 128 + nt * 16 + nLane;
    bcol[nt] = b[col];
    ccol[nt] = csum[col];
    #pragma unroll
    for (int r = 0; r < 8; ++r) {
      int row = rowBase + s * 16 + r + mHi;
      xc[nt][r] = x0[row * DIM + col];
    }
  }

  // Per-thread constant LDS bases (half-units); only immediates vary below.
  // Writer: element (m=r+mHi, col=q*128+nt*16+nLane) lands at plane nt&1,
  // k-tile q*4+(nt>>1), lane m+(nLane<8?0:16), h = nLane&7.
  const int aWr = s * 8192 + q * 2048 + mHi * 8 + (nLane & 7)
                + ((nLane < 8) ? 0 : 128);
  const int aRd = s * 8192 + lane * 8;       // + kt*512 (+256 for plane1)

  // seed(nt,r,t) = seedBase + const(nt,r) ^ tKey: one live register, the
  // per-element part folds into compile-time literals (no hoisted VGPR array).
  const uint32_t seedBase =
      (uint32_t)((rowBase + s * 16 + mHi) * DIM + q * 128 + nLane) * 0x9E3779B9u;

  const char* wbytes = (const char*)Wpk;

  #pragma unroll 1
  for (int t = 0; t < NUM_STEPS; ++t) {
    // --- 1) publish f16(BETA * x_t) into LDS (A-fragment order) ---
    #pragma unroll
    for (int nt = 0; nt < 8; ++nt) {
      #pragma unroll
      for (int r = 0; r < 8; ++r) {
        alds[aWr + (nt >> 1) * 512 + (nt & 1) * 256 + r * 8] =
            (_Float16)(BETA * xc[nt][r]);
      }
    }
    __syncthreads();

    // --- 2) xc becomes C_in = x_t + BETA*(b + t*csum) + SIGMA*noise ---
    float    tf   = (float)t;
    uint32_t tKey = (uint32_t)t * 0x85EBCA6Bu;
    #pragma unroll
    for (int nt = 0; nt < 8; ++nt) {
      float bt = BETA * (bcol[nt] + tf * ccol[nt]);
      #pragma unroll
      for (int r = 0; r < 8; ++r) {
        uint32_t seed =
            (seedBase + (uint32_t)(r * DIM + nt * 16) * 0x9E3779B9u) ^ tKey;
        xc[nt][r] += bt + SIGMA * gaussian(seed);
      }
    }

    // --- 3) xc = (BETA*x_t) @ W_top + C_in = x_{t+1}  (128 WMMAs/wave) ---
    // Opaque the W offset so B loads cannot be hoisted out of the t-loop.
    uint32_t wOff = (uint32_t)q * 131072u + (uint32_t)lane * 32u;
    asm volatile("" : "+v"(wOff));
    #pragma unroll
    for (int kt = 0; kt < 16; ++kt) {
      Frag A;
      A.q[0] = *(const uint4*)&alds[aRd + kt * 512];
      A.q[1] = *(const uint4*)&alds[aRd + kt * 512 + 256];
      #pragma unroll
      for (int nt = 0; nt < 8; ++nt) {
        Frag B;
        const char* bp = wbytes + wOff + (nt * 16384 + kt * 1024);
        B.q[0] = *(const uint4*)bp;
        B.q[1] = *(const uint4*)(bp + 16);
        xc[nt] = __builtin_amdgcn_wmma_f32_16x16x32_f16(
            false, A.v, false, B.v, (short)0, xc[nt], false, false);
      }
    }
    __syncthreads();
  }

  // --- write x_100 ---
  #pragma unroll
  for (int nt = 0; nt < 8; ++nt) {
    int col = q * 128 + nt * 16 + nLane;
    #pragma unroll
    for (int r = 0; r < 8; ++r) {
      int row = rowBase + s * 16 + r + mHi;
      out[row * DIM + col] = xc[nt][r];
    }
  }
}

extern "C" void kernel_launch(void* const* d_in, const int* in_sizes, int n_in,
                              void* d_out, int out_size, void* d_ws, size_t ws_size,
                              hipStream_t stream) {
  const float* x0 = (const float*)d_in[0];   // (32768, 512) f32
  const float* W  = (const float*)d_in[1];   // (1024, 512)  f32
  const float* b  = (const float*)d_in[2];   // (512,)       f32
  float* out = (float*)d_out;                // (32768, 512) f32

  _Float16* Wpk  = (_Float16*)d_ws;                                   // 512 KB
  float*    csum = (float*)((char*)d_ws + (size_t)DIM * KDIM * sizeof(_Float16));

  pack_w_kernel<<<(DIM * KDIM) / 256, 256, 0, stream>>>(W, Wpk);
  csum_kernel<<<(DIM + 255) / 256, 256, 0, stream>>>(W, csum);
  diffusion_kernel<<<BATCH / M_TILE, THREADS, 0, stream>>>(x0, Wpk, b, csum, out);
}
